// Qwen3Attention_26388279066841
// MI455X (gfx1250) — matrix operand order repaired and checked by reading, not compiled, not run on hardware
//
#include <hip/hip_runtime.h>
#include <hip/hip_bf16.h>

// ---------------- problem constants ----------------
#define BB   2
#define SS   2048
#define DD   2048
#define HH   16
#define KVH  8
#define HDIM 128
#define GRP  2
#define BSROWS (BB * SS)          // 4096
#define QKVN   (HH*HDIM + 2*KVH*HDIM)  // 4096
#define SCALE_F 0.08838834764831845f   // 128^-0.5
#define EPS_F   1e-6f
#define SP      2052               // padded LDS row stride (floats), 8B aligned, bank-spread

typedef __attribute__((ext_vector_type(16))) __bf16 v16bf;
typedef __attribute__((ext_vector_type(8)))  __bf16 v8bf;
typedef __attribute__((ext_vector_type(8)))  float  v8f;

// ---------------- WMMA helper ----------------
__device__ __forceinline__ v8f wmma_bf16(v16bf a, v16bf b, v8f c) {
  // D = A(16x32) * B(32x16) + C, f32 accumulate
  return __builtin_amdgcn_wmma_f32_16x16x32_bf16(false, a, false, b, (short)0, c, false, false);
}

// A fragment: 16x32 bf16 from row-major A[lda], tile origin (m0,k0).
// lanes 0-15: row M=lane, K = k0+{0..7,16..23}; lanes 16-31: row M=lane-16, K = k0+{8..15,24..31}
__device__ __forceinline__ v16bf load_a_frag(const __bf16* __restrict__ A, int lda,
                                             int m0, int k0, int lane) {
  const int row = m0 + (lane & 15);
  const int kb  = k0 + ((lane >> 4) << 3);
  const v8bf* p0 = reinterpret_cast<const v8bf*>(A + (size_t)row * lda + kb);
  const v8bf* p1 = reinterpret_cast<const v8bf*>(A + (size_t)row * lda + kb + 16);
  v8bf x0 = *p0, x1 = *p1;
  v16bf a;
#pragma unroll
  for (int i = 0; i < 8; ++i) { a[i] = x0[i]; a[i + 8] = x1[i]; }
  return a;
}

// B fragment: 32x16 bf16 = Wᵀ tile, W row-major [N][ldw], origin (n0,k0).
// lane n = n0+(lane&15); lanes 0-15: K=k0+{0..7,16..23}, lanes 16-31: K=k0+{8..15,24..31}
__device__ __forceinline__ v16bf load_b_frag(const __bf16* __restrict__ W, int ldw,
                                             int n0, int k0, int lane) {
  const int col = n0 + (lane & 15);
  const int kb  = k0 + ((lane >> 4) << 3);
  const v8bf* p = reinterpret_cast<const v8bf*>(W + (size_t)col * ldw + kb);
  v8bf x0 = p[0], x1 = p[2];
  v16bf b;
#pragma unroll
  for (int i = 0; i < 8; ++i) { b[i] = x0[i]; b[i + 8] = x1[i]; }
  return b;
}

// C/D 16x16 f32 tile store: lanes 0-15 hold M=v, lanes 16-31 hold M=8+v, col = lane&15
__device__ __forceinline__ void store_c(float* __restrict__ C, int N, int m0, int n0,
                                        int lane, v8f acc) {
  const int hi  = lane >> 4;
  const int col = n0 + (lane & 15);
#pragma unroll
  for (int v = 0; v < 8; ++v)
    C[(size_t)(m0 + v + 8 * hi) * N + col] = acc[v];
}

// ---------------- kernel 1: f32 -> bf16 convert ----------------
__global__ void k_f32_to_bf16(const float* __restrict__ in, __bf16* __restrict__ out, int n) {
  int i = blockIdx.x * blockDim.x + threadIdx.x;
  if (i < n) out[i] = (__bf16)in[i];
}

// ---------------- kernel 2/5: bf16 WMMA GEMM  C[M,N] = A[M,K] * W[N,K]^T ----------------
// block = 256 threads = 8 waves arranged 2(M) x 4(N); each wave computes a 32x32 tile
// (2x2 register blocking: 2 A-frags + 2 B-frags -> 4 WMMAs per K-step; 16 FLOP/byte)
__global__ void k_gemm_bf16(const __bf16* __restrict__ A, const __bf16* __restrict__ W,
                            float* __restrict__ C, int M, int N, int K) {
  const int lane = threadIdx.x & 31;
  const int wave = threadIdx.x >> 5;
  const int m0 = blockIdx.y * 64  + (wave & 1) * 32;   // wave's 32-row strip
  const int n0 = blockIdx.x * 128 + (wave >> 1) * 32;  // wave's 32-col strip
  v8f acc00 = {}, acc01 = {}, acc10 = {}, acc11 = {};
  for (int k0 = 0; k0 < K; k0 += 32) {
    v16bf a0 = load_a_frag(A, K, m0,      k0, lane);
    v16bf a1 = load_a_frag(A, K, m0 + 16, k0, lane);
    v16bf b0 = load_b_frag(W, K, n0,      k0, lane);
    v16bf b1 = load_b_frag(W, K, n0 + 16, k0, lane);
    acc00 = wmma_bf16(a0, b0, acc00);
    acc01 = wmma_bf16(a0, b1, acc01);
    acc10 = wmma_bf16(a1, b0, acc10);
    acc11 = wmma_bf16(a1, b1, acc11);
  }
  store_c(C, N, m0,      n0,      lane, acc00);
  store_c(C, N, m0,      n0 + 16, lane, acc01);
  store_c(C, N, m0 + 16, n0,      lane, acc10);
  store_c(C, N, m0 + 16, n0 + 16, lane, acc11);
}

// ---------------- kernel 3: RMSNorm + RoPE + layout shuffle ----------------
// grid = BSROWS * 32 blocks of 128 threads. slot 0-15: Q head; 16-23: K head; 24-31: V head.
__global__ void k_normrope(const float* __restrict__ qkv, const float* __restrict__ cosp,
                           const float* __restrict__ sinp, const float* __restrict__ qnw,
                           const float* __restrict__ knw,
                           __bf16* __restrict__ qh,   // (B,H,S,HD)
                           __bf16* __restrict__ kh,   // (B,KV,S,HD)
                           __bf16* __restrict__ vt) { // (B,KV,HD,S) transposed
  const int t    = threadIdx.x;           // 0..127
  const int slot = blockIdx.x & 31;
  const int row  = blockIdx.x >> 5;       // 0..4095
  const int b    = row >> 11;
  const int s    = row & (SS - 1);

  if (slot >= 24) {                       // V path: pure transpose, no norm/rope
    const int kv = slot - 24;
    float v = qkv[(size_t)row * QKVN + 3072 + kv * HDIM + t];
    vt[((size_t)(b * KVH + kv) * HDIM + t) * SS + s] = (__bf16)v;
    return;
  }

  __shared__ float xrow[HDIM];
  __shared__ float rbuf[HDIM];

  const bool isQ = slot < 16;
  const int  col = isQ ? (slot * HDIM + t) : (2048 + (slot - 16) * HDIM + t);
  const float x  = qkv[(size_t)row * QKVN + col];

  rbuf[t] = x * x;
  __syncthreads();
#pragma unroll
  for (int st = 64; st > 0; st >>= 1) {
    if (t < st) rbuf[t] += rbuf[t + st];
    __syncthreads();
  }
  const float rinv = rsqrtf(rbuf[0] * (1.0f / HDIM) + EPS_F);
  const float w    = isQ ? qnw[t] : knw[t];
  const float xn   = w * x * rinv;
  xrow[t] = xn;
  __syncthreads();

  const float rot = (t < 64) ? -xrow[t + 64] : xrow[t - 64];
  const float c   = cosp[((size_t)b * SS + s) * HDIM + t];
  const float sn  = sinp[((size_t)b * SS + s) * HDIM + t];
  const float out = xn * c + rot * sn;

  if (isQ) {
    qh[((size_t)(b * HH + slot) * SS + s) * HDIM + t] = (__bf16)out;
  } else {
    const int kv = slot - 16;
    kh[((size_t)(b * KVH + kv) * SS + s) * HDIM + t] = (__bf16)out;
  }
}

// ---------------- kernel 4: attention (scores -> softmax -> probs out -> P@V) ----------
// one block (256 thr / 8 waves) per (b, h, 16-row query tile); dynamic LDS 16 x SP f32
__global__ void k_attention(const __bf16* __restrict__ qh, const __bf16* __restrict__ kh,
                            const __bf16* __restrict__ vt, const float* __restrict__ mask,
                            float* __restrict__ attn_out,   // (B,H,S,S) f32
                            __bf16* __restrict__ ctxb) {    // (B,S,H*HD) bf16
  extern __shared__ float smem[];
  float* scp     = smem;                 // 16 * SP
  float* red     = smem + 16 * SP;       // 256
  float* rowstat = red + 256;            // 16

  const int tid  = threadIdx.x;
  const int lane = tid & 31;
  const int wave = tid >> 5;

  const int qt  = blockIdx.x & 127;
  const int h   = (blockIdx.x >> 7) & 15;
  const int b   = blockIdx.x >> 11;
  const int q0  = qt * 16;
  const int kv  = h / GRP;

  // preload Q tile fragments (16 x 128)
  const __bf16* qbase = qh + ((size_t)(b * HH + h) * SS + q0) * HDIM;
  v16bf aq[4];
#pragma unroll
  for (int c = 0; c < 4; ++c) aq[c] = load_a_frag(qbase, HDIM, 0, c * 32, lane);

  // ---- phase 1: scores = Q K^T * scale + mask -> LDS ----
  const __bf16* kbase = kh + (size_t)(b * KVH + kv) * SS * HDIM; // rows = t, ld = HD
  for (int j = wave; j < SS / 16; j += 8) {
    const int t0 = j * 16;
    v8f acc = {};
#pragma unroll
    for (int c = 0; c < 4; ++c)
      acc = wmma_bf16(aq[c], load_b_frag(kbase, HDIM, t0, c * 32, lane), acc);
    const int hi = lane >> 4;
    const int tc = t0 + (lane & 15);
#pragma unroll
    for (int v = 0; v < 8; ++v) {
      const int r = v + 8 * hi;
      float val = acc[v] * SCALE_F + mask[((size_t)b * SS + (q0 + r)) * SS + tc];
      scp[r * SP + tc] = val;
    }
  }
  __syncthreads();

  // ---- phase 2: softmax over each of 16 rows (16 threads per row) ----
  const int r   = tid >> 4;
  const int sub = tid & 15;
  float lmax = -1e30f;
#pragma unroll 4
  for (int i = 0; i < 128; ++i)
    lmax = fmaxf(lmax, scp[r * SP + sub * 128 + i]);
  red[tid] = lmax;
  __syncthreads();
  if (sub == 0) {
    float m = red[r * 16];
#pragma unroll
    for (int i = 1; i < 16; ++i) m = fmaxf(m, red[r * 16 + i]);
    rowstat[r] = m;
  }
  __syncthreads();
  const float rmax = rowstat[r];
  float lsum = 0.f;
#pragma unroll 4
  for (int i = 0; i < 128; ++i) {
    float e = __expf(scp[r * SP + sub * 128 + i] - rmax);
    scp[r * SP + sub * 128 + i] = e;
    lsum += e;
  }
  red[tid] = lsum;
  __syncthreads();
  if (sub == 0) {
    float s = red[r * 16];
#pragma unroll
    for (int i = 1; i < 16; ++i) s += red[r * 16 + i];
    rowstat[r] = 1.0f / s;
  }
  __syncthreads();

  // normalize + coalesced write of probabilities
  float* aout = attn_out + ((size_t)(b * HH + h) * SS + q0) * SS;
  for (int idx = tid; idx < 16 * SS; idx += 256) {
    const int rr = idx >> 11, cc = idx & (SS - 1);
    float v = scp[rr * SP + cc] * rowstat[rr];
    scp[rr * SP + cc] = v;
    aout[(size_t)rr * SS + cc] = v;
  }
  __syncthreads();

  // ---- phase 3: ctx = P @ V ; wave w owns hd-tile n0 = w*16 ----
  const int n0 = wave * 16;
  const __bf16* vbase = vt + (size_t)(b * KVH + kv) * HDIM * SS; // rows = hd, ld = S
  v8f acc = {};
  for (int j = 0; j < SS / 32; ++j) {
    const int t0 = j * 32;
    // A fragment: P rows from LDS (f32 -> bf16)
    const int m  = lane & 15;
    const int kb = t0 + ((lane >> 4) << 3);
    v16bf ap;
#pragma unroll
    for (int v = 0; v < 4; ++v) {
      float2 f0 = *reinterpret_cast<const float2*>(scp + m * SP + kb + 2 * v);
      float2 f1 = *reinterpret_cast<const float2*>(scp + m * SP + kb + 16 + 2 * v);
      ap[2 * v]     = (__bf16)f0.x;  ap[2 * v + 1] = (__bf16)f0.y;
      ap[8 + 2 * v] = (__bf16)f1.x;  ap[9 + 2 * v] = (__bf16)f1.y;
    }
    v16bf bv = load_b_frag(vbase, SS, n0, t0, lane);
    acc = wmma_bf16(ap, bv, acc);
  }
  const int hi = lane >> 4;
  const int nc = n0 + (lane & 15);
#pragma unroll
  for (int v = 0; v < 8; ++v) {
    const int qrow = q0 + v + 8 * hi;
    ctxb[((size_t)(b * SS + qrow)) * (HH * HDIM) + h * HDIM + nc] = (__bf16)acc[v];
  }
}

// ---------------- host launcher ----------------
extern "C" void kernel_launch(void* const* d_in, const int* in_sizes, int n_in,
                              void* d_out, int out_size, void* d_ws, size_t ws_size,
                              hipStream_t stream) {
  (void)in_sizes; (void)n_in; (void)out_size; (void)ws_size;
  const float* x    = (const float*)d_in[0];
  const float* cosp = (const float*)d_in[1];
  const float* sinp = (const float*)d_in[2];
  const float* mask = (const float*)d_in[3];
  const float* wq   = (const float*)d_in[4];
  const float* wk   = (const float*)d_in[5];
  const float* wv   = (const float*)d_in[6];
  const float* wo   = (const float*)d_in[7];
  const float* qnw  = (const float*)d_in[8];
  const float* knw  = (const float*)d_in[9];

  float* out      = (float*)d_out;                        // (B,S,D)
  float* out_attn = out + (size_t)BB * SS * DD;           // (B,H,S,S)

  // workspace carve-up
  char* ws = (char*)d_ws;
  size_t off = 0;
  auto carve = [&](size_t bytes) { void* p = ws + off; off += (bytes + 255) & ~(size_t)255; return p; };
  __bf16* xb    = (__bf16*)carve((size_t)BSROWS * DD * 2);            // 16 MB
  __bf16* wqkvb = (__bf16*)carve((size_t)QKVN * DD * 2);              // 16 MB
  __bf16* wob   = (__bf16*)carve((size_t)DD * (HH * HDIM) * 2);       // 8 MB
  float*  qkv   = (float*)carve((size_t)BSROWS * QKVN * 4);           // 64 MB
  __bf16* qh    = (__bf16*)carve((size_t)BB * HH * SS * HDIM * 2);    // 16 MB
  __bf16* kh    = (__bf16*)carve((size_t)BB * KVH * SS * HDIM * 2);   // 8 MB
  __bf16* vt    = (__bf16*)carve((size_t)BB * KVH * HDIM * SS * 2);   // 8 MB
  __bf16* ctxb  = (__bf16*)carve((size_t)BSROWS * HH * HDIM * 2);     // 16 MB

  const int CT = 256;
  auto cvgrid = [](int n) { return (n + 255) / 256; };
  k_f32_to_bf16<<<cvgrid(BSROWS * DD), CT, 0, stream>>>(x, xb, BSROWS * DD);
  k_f32_to_bf16<<<cvgrid(HH * HDIM * DD), CT, 0, stream>>>(wq, wqkvb, HH * HDIM * DD);
  k_f32_to_bf16<<<cvgrid(KVH * HDIM * DD), CT, 0, stream>>>(wk, wqkvb + (size_t)HH * HDIM * DD, KVH * HDIM * DD);
  k_f32_to_bf16<<<cvgrid(KVH * HDIM * DD), CT, 0, stream>>>(wv, wqkvb + (size_t)(HH + KVH) * HDIM * DD, KVH * HDIM * DD);
  k_f32_to_bf16<<<cvgrid(DD * HH * HDIM), CT, 0, stream>>>(wo, wob, DD * HH * HDIM);

  // qkv = x @ [wq;wk;wv]^T   (block tile 64M x 128N, wave tile 32x32)
  k_gemm_bf16<<<dim3(QKVN / 128, BSROWS / 64), 256, 0, stream>>>(xb, wqkvb, qkv, BSROWS, QKVN, DD);

  // rmsnorm + rope + layout
  k_normrope<<<BSROWS * 32, 128, 0, stream>>>(qkv, cosp, sinp, qnw, knw, qh, kh, vt);

  // attention: 16x2048 f32 score tile + reductions in dynamic LDS
  const size_t shmem = (size_t)(16 * SP + 256 + 16) * sizeof(float);
  hipFuncSetAttribute((const void*)k_attention, hipFuncAttributeMaxDynamicSharedMemorySize, (int)shmem);
  k_attention<<<BB * HH * (SS / 16), 256, shmem, stream>>>(qh, kh, vt, mask, out_attn, ctxb);

  // out = ctx @ wo^T
  k_gemm_bf16<<<dim3(DD / 128, BSROWS / 64), 256, 0, stream>>>(ctxb, wob, out, BSROWS, DD, HH * HDIM);
}
